// SpatioTemporalBlock_5342939316375
// MI455X (gfx1250) — matrix operand order repaired and checked by reading, not compiled, not run on hardware
//
#include <hip/hip_runtime.h>

// ---------------------------------------------------------------------------
// SpatioTemporalBlock for MI455X (gfx1250, wave32, WMMA).
// Heavy math via v_wmma_f32_16x16x32_f16 (f16 in, f32 accumulate).
// GEMM tiles are staged with CDNA5 async global->LDS copies
// (global_load_async_to_lds_b128, ASYNCcnt) with double buffering.
// Weights are transposed+converted to f16 [N][K] once per call.
// Workspace requirement: ~295.7 MB (see layout in kernel_launch).
// ---------------------------------------------------------------------------

typedef __attribute__((ext_vector_type(16))) _Float16 v16h;
typedef __attribute__((ext_vector_type(8)))  _Float16 v8h;
typedef __attribute__((ext_vector_type(4)))  _Float16 v4h;
typedef __attribute__((ext_vector_type(8)))  float    v8f;

#define CAT16(lo, hi) __builtin_shufflevector(lo, hi, 0,1,2,3,4,5,6,7,8,9,10,11,12,13,14,15)

static __device__ __forceinline__ float redmax16(float v) {
#pragma unroll
  for (int o = 8; o >= 1; o >>= 1) v = fmaxf(v, __shfl_xor(v, o, 32));
  return v;
}
static __device__ __forceinline__ float redsum16(float v) {
#pragma unroll
  for (int o = 8; o >= 1; o >>= 1) v += __shfl_xor(v, o, 32);
  return v;
}

// Async copy of 16 bytes global -> LDS (per lane). VDST carries the LDS byte
// offset (low 32 bits of the generic pointer == LDS offset on gfx1250, since
// the SHARED aperture occupies addr[63:32]). Tracked by ASYNCcnt.
static __device__ __forceinline__ void async_copy_b128(unsigned lds_off,
                                                       const _Float16* gptr) {
  asm volatile("global_load_async_to_lds_b128 %0, %1, off"
               :: "v"(lds_off), "v"(gptr)
               : "memory");
}
static __device__ __forceinline__ unsigned lds_addr(const void* p) {
  return (unsigned)(unsigned long long)p;
}

// ---------------------------------------------------------------------------
// Weight prep: WT[n][k] = (f16) W[k][n], 32x32 LDS-tiled transpose.
// Grid: (N/32, K/32), 256 threads (32x8).
// ---------------------------------------------------------------------------
__global__ __launch_bounds__(256)
void transpose_w_f16(const float* __restrict__ W, _Float16* __restrict__ WT,
                     int K, int N) {
  __shared__ float t[32][33];
  const int kb = blockIdx.y * 32, nb = blockIdx.x * 32;
  const int tx = threadIdx.x & 31, ty = threadIdx.x >> 5;
#pragma unroll
  for (int i = 0; i < 4; ++i)
    t[ty + 8 * i][tx] = W[(long)(kb + ty + 8 * i) * N + nb + tx];
  __syncthreads();
#pragma unroll
  for (int i = 0; i < 4; ++i)
    WT[(long)(nb + ty + 8 * i) * K + kb + tx] = (_Float16)t[tx][ty + 8 * i];
}

// f32 -> f16 bulk convert; each thread handles 4 elements (n divisible by 1024)
__global__ __launch_bounds__(256)
void cvt_f16(const float* __restrict__ in, _Float16* __restrict__ out) {
  const long i = ((long)blockIdx.x * 256 + threadIdx.x) * 4;
  const float4 a = *(const float4*)(in + i);
  v4h h = {(_Float16)a.x, (_Float16)a.y, (_Float16)a.z, (_Float16)a.w};
  *(v4h*)(out + i) = h;
}

// ---------------------------------------------------------------------------
// WMMA GEMM: out = epilogue(A[MxK](f16) @ BT[NxK](f16)^T + bias [+ resid])
// Block 256 threads (8 waves), tile 128x128, BK=64 (2 WMMA K-steps / stage).
// Wave grid 2x4; each wave computes 64x32 -> 16 WMMAs per stage.
// Tiles staged with double-buffered async global->LDS b128 copies
// (8 per thread per stage), synchronized via s_wait_asynccnt + barrier.
// ---------------------------------------------------------------------------
template <bool GELU_ACT, bool RESID, bool OUT_HALF>
__global__ __launch_bounds__(256)
void gemm_wmma(const _Float16* __restrict__ A, const _Float16* __restrict__ BT,
               const float* __restrict__ bias, const float* __restrict__ resid,
               void* __restrict__ outv, int M, int N, int K) {
  __shared__ _Float16 sA[2][128][72];
  __shared__ _Float16 sB[2][128][72];   // sB[buf][n][k]

  const int tid  = threadIdx.x;
  const int lane = tid & 31;
  const int wv   = tid >> 5;
  const int lm = lane & 15, lg = lane >> 4;
  const int wm = wv >> 2,  wn = wv & 3;
  const long m0 = (long)blockIdx.y * 128;
  const long n0 = (long)blockIdx.x * 128;

  const v8f vzero = {0.f, 0.f, 0.f, 0.f, 0.f, 0.f, 0.f, 0.f};
  v8f acc[4][2];
#pragma unroll
  for (int i = 0; i < 4; ++i) { acc[i][0] = vzero; acc[i][1] = vzero; }

  const int row = tid >> 1;          // 0..127
  const int cb  = (tid & 1) * 32;    // half-row of the 64-wide K slab

  auto issue_stage = [&](int buf, int k0) {
    const _Float16* ga = A  + (m0 + row) * (long)K + k0 + cb;
    const _Float16* gb = BT + (n0 + row) * (long)K + k0 + cb;
    const unsigned la = lds_addr(&sA[buf][row][cb]);
    const unsigned lb = lds_addr(&sB[buf][row][cb]);
    async_copy_b128(la,      ga);
    async_copy_b128(la + 16, ga + 8);
    async_copy_b128(la + 32, ga + 16);
    async_copy_b128(la + 48, ga + 24);
    async_copy_b128(lb,      gb);
    async_copy_b128(lb + 16, gb + 8);
    async_copy_b128(lb + 32, gb + 16);
    async_copy_b128(lb + 48, gb + 24);
  };

  issue_stage(0, 0);
  int cur = 0;
  for (int k0 = 0; k0 < K; k0 += 64) {
    const bool has_next = (k0 + 64) < K;
    if (has_next) {
      issue_stage(cur ^ 1, k0 + 64);
      // 8 copies in flight for next buffer; in-order completion guarantees
      // the current buffer's 8 copies are done once count drops to 8.
      asm volatile("s_wait_asynccnt 0x8" ::: "memory");
    } else {
      asm volatile("s_wait_asynccnt 0x0" ::: "memory");
    }
    __syncthreads();

#pragma unroll
    for (int ks = 0; ks < 2; ++ks) {
      v16h af[4];
#pragma unroll
      for (int mt = 0; mt < 4; ++mt) {
        const int r = wm * 64 + mt * 16 + lm;
        v8h lo = *(const v8h*)&sA[cur][r][ks * 32 + 16 * lg];
        v8h hi = *(const v8h*)&sA[cur][r][ks * 32 + 16 * lg + 8];
        af[mt] = CAT16(lo, hi);
      }
      v16h bf[2];
#pragma unroll
      for (int nt = 0; nt < 2; ++nt) {
        const int c = wn * 32 + nt * 16 + lm;
        v8h lo = *(const v8h*)&sB[cur][c][ks * 32 + 16 * lg];
        v8h hi = *(const v8h*)&sB[cur][c][ks * 32 + 16 * lg + 8];
        bf[nt] = CAT16(lo, hi);
      }
#pragma unroll
      for (int mt = 0; mt < 4; ++mt)
#pragma unroll
        for (int nt = 0; nt < 2; ++nt)
          acc[mt][nt] = __builtin_amdgcn_wmma_f32_16x16x32_f16(
              false, af[mt], false, bf[nt], (short)0, acc[mt][nt], false, false);
    }
    __syncthreads();   // all waves done reading `cur` before it is re-filled
    cur ^= 1;
  }

#pragma unroll
  for (int mt = 0; mt < 4; ++mt)
#pragma unroll
    for (int nt = 0; nt < 2; ++nt)
#pragma unroll
      for (int r = 0; r < 8; ++r) {
        const long gm = m0 + wm * 64 + mt * 16 + 8 * lg + r;
        const long gn = n0 + wn * 32 + nt * 16 + lm;
        float x = acc[mt][nt][r] + bias[gn];
        if constexpr (RESID)    x += resid[gm * N + gn];
        if constexpr (GELU_ACT) x = 0.5f * x * (1.0f + erff(x * 0.70710678118654752f));
        if constexpr (OUT_HALF) ((_Float16*)outv)[gm * N + gn] = (_Float16)x;
        else                    ((float*)outv)[gm * N + gn] = x;
      }
}

// ---------------------------------------------------------------------------
// Spatial attention: one block per (b*t, head). P=256 keys, D=64.
// ---------------------------------------------------------------------------
__global__ __launch_bounds__(256)
void spatial_attn(const _Float16* __restrict__ q, const _Float16* __restrict__ k,
                  const _Float16* __restrict__ v, _Float16* __restrict__ out) {
  __shared__ _Float16 sK[256][72];    // [key][d]
  __shared__ _Float16 sVT[64][264];   // [d][key]
  __shared__ _Float16 sP[8][32][40];  // per-wave probability patch

  const int tid  = threadIdx.x;
  const int lane = tid & 31, w = tid >> 5;
  const int lm = lane & 15, lg = lane >> 4;
  const int bt = blockIdx.x / 12;
  const int h  = blockIdx.x % 12;
  const long base = (long)bt * 256 * 768 + h * 64;

  {  // stage K row-major and V transposed
    const v8h* kp = (const v8h*)(k + base + (long)tid * 768);
#pragma unroll
    for (int i = 0; i < 8; ++i) *(v8h*)&sK[tid][i * 8] = kp[i];
    const v8h* vp = (const v8h*)(v + base + (long)tid * 768);
#pragma unroll
    for (int i = 0; i < 8; ++i) {
      v8h vr = vp[i];
#pragma unroll
      for (int j = 0; j < 8; ++j) sVT[i * 8 + j][tid] = vr[j];
    }
  }
  __syncthreads();

  v16h qf[2][2];
#pragma unroll
  for (int mt = 0; mt < 2; ++mt)
#pragma unroll
    for (int ks = 0; ks < 2; ++ks) {
      const _Float16* qp = q + base + (long)(w * 32 + mt * 16 + lm) * 768 + ks * 32;
      v8h lo = *(const v8h*)(qp + 16 * lg);
      v8h hi = *(const v8h*)(qp + 16 * lg + 8);
      qf[mt][ks] = CAT16(lo, hi);
    }

  const v8f vzero = {0.f, 0.f, 0.f, 0.f, 0.f, 0.f, 0.f, 0.f};
  v8f s[2][16];
#pragma unroll
  for (int mt = 0; mt < 2; ++mt)
#pragma unroll
    for (int nt = 0; nt < 16; ++nt) s[mt][nt] = vzero;

  // S = Q @ K^T
#pragma unroll
  for (int nt = 0; nt < 16; ++nt) {
    v16h bf[2];
#pragma unroll
    for (int ks = 0; ks < 2; ++ks) {
      const int kn = nt * 16 + lm;
      v8h lo = *(const v8h*)&sK[kn][ks * 32 + 16 * lg];
      v8h hi = *(const v8h*)&sK[kn][ks * 32 + 16 * lg + 8];
      bf[ks] = CAT16(lo, hi);
    }
#pragma unroll
    for (int mt = 0; mt < 2; ++mt)
#pragma unroll
      for (int ks = 0; ks < 2; ++ks)
        s[mt][nt] = __builtin_amdgcn_wmma_f32_16x16x32_f16(
            false, qf[mt][ks], false, bf[ks], (short)0, s[mt][nt], false, false);
  }

  // row softmax (scale 1/sqrt(64) = 0.125)
#pragma unroll
  for (int mt = 0; mt < 2; ++mt)
#pragma unroll
    for (int r = 0; r < 8; ++r) {
      float mx = -3.0e38f;
#pragma unroll
      for (int nt = 0; nt < 16; ++nt) mx = fmaxf(mx, s[mt][nt][r]);
      mx = redmax16(mx);
      float sum = 0.f;
#pragma unroll
      for (int nt = 0; nt < 16; ++nt) {
        float e = __expf((s[mt][nt][r] - mx) * 0.125f);
        s[mt][nt][r] = e;
        sum += e;
      }
      sum = redsum16(sum);
      const float inv = 1.f / sum;
#pragma unroll
      for (int nt = 0; nt < 16; ++nt) s[mt][nt][r] *= inv;
    }

  // O = P @ V
  v8f o[2][4];
#pragma unroll
  for (int mt = 0; mt < 2; ++mt)
#pragma unroll
    for (int dt = 0; dt < 4; ++dt) o[mt][dt] = vzero;

  for (int ch = 0; ch < 8; ++ch) {
#pragma unroll
    for (int mt = 0; mt < 2; ++mt)
#pragma unroll
      for (int nt2 = 0; nt2 < 2; ++nt2) {
        const int nt = ch * 2 + nt2;
#pragma unroll
        for (int r = 0; r < 8; ++r)
          sP[w][mt * 16 + 8 * lg + r][nt2 * 16 + lm] = (_Float16)s[mt][nt][r];
      }
    __syncthreads();
    v16h pf[2];
#pragma unroll
    for (int mt = 0; mt < 2; ++mt) {
      v8h lo = *(const v8h*)&sP[w][mt * 16 + lm][16 * lg];
      v8h hi = *(const v8h*)&sP[w][mt * 16 + lm][16 * lg + 8];
      pf[mt] = CAT16(lo, hi);
    }
    v16h vf[4];
#pragma unroll
    for (int dt = 0; dt < 4; ++dt) {
      const int dn = dt * 16 + lm;
      v8h lo = *(const v8h*)&sVT[dn][ch * 32 + 16 * lg];
      v8h hi = *(const v8h*)&sVT[dn][ch * 32 + 16 * lg + 8];
      vf[dt] = CAT16(lo, hi);
    }
#pragma unroll
    for (int mt = 0; mt < 2; ++mt)
#pragma unroll
      for (int dt = 0; dt < 4; ++dt)
        o[mt][dt] = __builtin_amdgcn_wmma_f32_16x16x32_f16(
            false, pf[mt], false, vf[dt], (short)0, o[mt][dt], false, false);
    __syncthreads();
  }

#pragma unroll
  for (int mt = 0; mt < 2; ++mt)
#pragma unroll
    for (int dt = 0; dt < 4; ++dt)
#pragma unroll
      for (int r = 0; r < 8; ++r)
        out[base + (long)(w * 32 + mt * 16 + 8 * lg + r) * 768 + dt * 16 + lm] =
            (_Float16)o[mt][dt][r];
}

// ---------------------------------------------------------------------------
// Temporal causal attention: one wave per (b, p, head); 16x16 = one WMMA tile.
// ---------------------------------------------------------------------------
__global__ __launch_bounds__(256)
void temporal_attn(const _Float16* __restrict__ q, const _Float16* __restrict__ k,
                   const _Float16* __restrict__ v, _Float16* __restrict__ out) {
  __shared__ _Float16 sTP[8][16][40];

  const int tid  = threadIdx.x;
  const int lane = tid & 31, w = tid >> 5;
  const int lm = lane & 15, lg = lane >> 4;
  const int gw = blockIdx.x * 8 + w;          // 0 .. B*P*H-1
  const int h  = gw % 12;
  const int bp = gw / 12;
  const int p  = bp & 255;
  const int b  = bp >> 8;
  const long base = ((long)b * 4096 + p) * 768 + h * 64;
  const long ts = 196608;                     // P*768 halves per t-step

  v16h qf[2], kf[2];
#pragma unroll
  for (int ks = 0; ks < 2; ++ks) {
    const _Float16* qp = q + base + (long)lm * ts + ks * 32;
    qf[ks] = CAT16(*(const v8h*)(qp + 16 * lg), *(const v8h*)(qp + 16 * lg + 8));
    const _Float16* kp = k + base + (long)lm * ts + ks * 32 + 16 * lg;
    kf[ks] = CAT16(*(const v8h*)kp, *(const v8h*)(kp + 8));
  }

  const v8f vzero = {0.f, 0.f, 0.f, 0.f, 0.f, 0.f, 0.f, 0.f};
  v8f s = vzero;
#pragma unroll
  for (int ks = 0; ks < 2; ++ks)
    s = __builtin_amdgcn_wmma_f32_16x16x32_f16(false, qf[ks], false, kf[ks],
                                               (short)0, s, false, false);

  // causal mask + softmax; C layout: row m = 8*lg + r, col n = lm
#pragma unroll
  for (int r = 0; r < 8; ++r) {
    const int m = 8 * lg + r;
    float val = (lm <= m) ? s[r] * 0.125f : -3.0e38f;
    float mx = redmax16(val);
    float e = __expf(val - mx);
    float sum = redsum16(e);
    s[r] = e / sum;
  }

#pragma unroll
  for (int r = 0; r < 8; ++r) {
    sTP[w][8 * lg + r][lm]      = (_Float16)s[r];
    sTP[w][8 * lg + r][16 + lm] = (_Float16)0.f;
  }
  __syncthreads();
  v16h pf = CAT16(*(const v8h*)&sTP[w][lm][16 * lg],
                  *(const v8h*)&sTP[w][lm][16 * lg + 8]);

  v8f o[4];
#pragma unroll
  for (int dt = 0; dt < 4; ++dt) o[dt] = vzero;
#pragma unroll
  for (int dt = 0; dt < 4; ++dt) {
    v16h vf;
#pragma unroll
    for (int j = 0; j < 16; ++j) {
      _Float16 x = (lg == 0) ? v[base + (long)j * ts + dt * 16 + lm] : (_Float16)0.f;
      vf[j] = x;
    }
    o[dt] = __builtin_amdgcn_wmma_f32_16x16x32_f16(false, pf, false, vf,
                                                   (short)0, o[dt], false, false);
  }

#pragma unroll
  for (int dt = 0; dt < 4; ++dt)
#pragma unroll
    for (int r = 0; r < 8; ++r)
      out[base + (long)(8 * lg + r) * ts + dt * 16 + lm] = (_Float16)o[dt][r];
}

// ---------------------------------------------------------------------------
// Row LayerNorm over E=768 (f32); optionally emits an f16 copy for the next
// GEMM's A-side. One block per row, 3 elements per thread.
// ---------------------------------------------------------------------------
template <bool EMIT_HALF>
__global__ __launch_bounds__(256)
void ln_kernel(const float* __restrict__ in, const float* __restrict__ g,
               const float* __restrict__ b, float* __restrict__ out,
               _Float16* __restrict__ outh) {
  __shared__ float red[8];
  const long row = blockIdx.x;
  const float* x = in + row * 768;
  float* o = out + row * 768;
  const int tid = threadIdx.x;
  const int w = tid >> 5, lane = tid & 31;

  float a0 = x[tid], a1 = x[tid + 256], a2 = x[tid + 512];
  float s = a0 + a1 + a2;
#pragma unroll
  for (int off = 16; off >= 1; off >>= 1) s += __shfl_xor(s, off, 32);
  if (lane == 0) red[w] = s;
  __syncthreads();
  float tot = 0.f;
#pragma unroll
  for (int i = 0; i < 8; ++i) tot += red[i];
  const float mean = tot * (1.0f / 768.0f);

  float d0 = a0 - mean, d1 = a1 - mean, d2 = a2 - mean;
  float qs = d0 * d0 + d1 * d1 + d2 * d2;
#pragma unroll
  for (int off = 16; off >= 1; off >>= 1) qs += __shfl_xor(qs, off, 32);
  __syncthreads();
  if (lane == 0) red[w] = qs;
  __syncthreads();
  float tv = 0.f;
#pragma unroll
  for (int i = 0; i < 8; ++i) tv += red[i];
  const float rstd = rsqrtf(tv * (1.0f / 768.0f) + 1e-5f);

  const float r0 = d0 * rstd * g[tid]       + b[tid];
  const float r1 = d1 * rstd * g[tid + 256] + b[tid + 256];
  const float r2 = d2 * rstd * g[tid + 512] + b[tid + 512];
  o[tid] = r0; o[tid + 256] = r1; o[tid + 512] = r2;
  if constexpr (EMIT_HALF) {
    _Float16* oh = outh + row * 768;
    oh[tid] = (_Float16)r0; oh[tid + 256] = (_Float16)r1; oh[tid + 512] = (_Float16)r2;
  }
}

// ---------------------------------------------------------------------------
extern "C" void kernel_launch(void* const* d_in, const int* in_sizes, int n_in,
                              void* d_out, int out_size, void* d_ws, size_t ws_size,
                              hipStream_t stream) {
  (void)in_sizes; (void)n_in; (void)out_size; (void)ws_size;

  const float* x    = (const float*)d_in[0];
  const float* sq_w = (const float*)d_in[1];
  const float* sk_w = (const float*)d_in[2];
  const float* sv_w = (const float*)d_in[3];
  const float* so_w = (const float*)d_in[4];
  const float* sq_b = (const float*)d_in[5];
  const float* sk_b = (const float*)d_in[6];
  const float* sv_b = (const float*)d_in[7];
  const float* so_b = (const float*)d_in[8];
  const float* sg   = (const float*)d_in[9];
  const float* sb   = (const float*)d_in[10];
  const float* tq_w = (const float*)d_in[11];
  const float* tk_w = (const float*)d_in[12];
  const float* tv_w = (const float*)d_in[13];
  const float* to_w = (const float*)d_in[14];
  const float* tq_b = (const float*)d_in[15];
  const float* tk_b = (const float*)d_in[16];
  const float* tv_b = (const float*)d_in[17];
  const float* to_b = (const float*)d_in[18];
  const float* tg   = (const float*)d_in[19];
  const float* tb   = (const float*)d_in[20];
  const float* w1   = (const float*)d_in[21];
  const float* b1   = (const float*)d_in[22];
  const float* w2   = (const float*)d_in[23];
  const float* b2   = (const float*)d_in[24];
  const float* g    = (const float*)d_in[25];
  const float* bb   = (const float*)d_in[26];

  // ---- workspace layout (bytes) ----
  char* ws = (char*)d_ws;
  const long QS = 25165824L;     // 16384*768 f16
  const long FS = 50331648L;     // 16384*768 f32
  const long WT768 = 1179648L;   // 768*768  f16
  const long WTBIG = 4718592L;   // 768*3072 f16
  _Float16* q    = (_Float16*)(ws);
  _Float16* kk   = (_Float16*)(ws + QS);
  _Float16* vv   = (_Float16*)(ws + 2 * QS);
  _Float16* attn = (_Float16*)(ws + 3 * QS);
  _Float16* hbuf = (_Float16*)(ws);               // reuses q..attn (= M*3072 f16)
  float* tmp = (float*)(ws + 4 * QS);
  float* x1  = (float*)(ws + 4 * QS + FS);
  float* x2  = (float*)(ws + 4 * QS + 2 * FS);
  _Float16* ah = (_Float16*)(ws + 4 * QS + 3 * FS); // xh / x1h / x2h (reused)
  char* wbase = ws + 4 * QS + 3 * FS + QS;
  _Float16* sqT = (_Float16*)(wbase + 0 * WT768);
  _Float16* skT = (_Float16*)(wbase + 1 * WT768);
  _Float16* svT = (_Float16*)(wbase + 2 * WT768);
  _Float16* soT = (_Float16*)(wbase + 3 * WT768);
  _Float16* tqT = (_Float16*)(wbase + 4 * WT768);
  _Float16* tkT = (_Float16*)(wbase + 5 * WT768);
  _Float16* tvT = (_Float16*)(wbase + 6 * WT768);
  _Float16* toT = (_Float16*)(wbase + 7 * WT768);
  _Float16* w1T = (_Float16*)(wbase + 8 * WT768);
  _Float16* w2T = (_Float16*)(wbase + 8 * WT768 + WTBIG);
  // total = 295,698,432 bytes

  const int M = 16384, E = 768, DFF = 3072;
  dim3 blk(256);
  dim3 gE(E / 128, M / 128);     // 6 x 128
  dim3 gF(DFF / 128, M / 128);   // 24 x 128
  dim3 gT(E / 32, E / 32);       // 24 x 24
  dim3 gT1(DFF / 32, E / 32);    // 96 x 24  (w1: K=E, N=DFF)
  dim3 gT2(E / 32, DFF / 32);    // 24 x 96  (w2: K=DFF, N=E)

  // ---- one-time per-call prep: weights -> f16 [N][K]; x -> f16 ----
  transpose_w_f16<<<gT,  blk, 0, stream>>>(sq_w, sqT, E, E);
  transpose_w_f16<<<gT,  blk, 0, stream>>>(sk_w, skT, E, E);
  transpose_w_f16<<<gT,  blk, 0, stream>>>(sv_w, svT, E, E);
  transpose_w_f16<<<gT,  blk, 0, stream>>>(so_w, soT, E, E);
  transpose_w_f16<<<gT,  blk, 0, stream>>>(tq_w, tqT, E, E);
  transpose_w_f16<<<gT,  blk, 0, stream>>>(tk_w, tkT, E, E);
  transpose_w_f16<<<gT,  blk, 0, stream>>>(tv_w, tvT, E, E);
  transpose_w_f16<<<gT,  blk, 0, stream>>>(to_w, toT, E, E);
  transpose_w_f16<<<gT1, blk, 0, stream>>>(w1, w1T, E, DFF);
  transpose_w_f16<<<gT2, blk, 0, stream>>>(w2, w2T, DFF, E);
  cvt_f16<<<dim3(12288), blk, 0, stream>>>(x, ah);

  // ---- Spatial layer ----
  gemm_wmma<false, false, true><<<gE, blk, 0, stream>>>(ah, sqT, sq_b, nullptr, q,  M, E, E);
  gemm_wmma<false, false, true><<<gE, blk, 0, stream>>>(ah, skT, sk_b, nullptr, kk, M, E, E);
  gemm_wmma<false, false, true><<<gE, blk, 0, stream>>>(ah, svT, sv_b, nullptr, vv, M, E, E);
  spatial_attn<<<dim3(768), blk, 0, stream>>>(q, kk, vv, attn);
  gemm_wmma<false, true, false><<<gE, blk, 0, stream>>>(attn, soT, so_b, x, tmp, M, E, E);
  ln_kernel<true><<<dim3(16384), blk, 0, stream>>>(tmp, sg, sb, x1, ah);

  // ---- Temporal layer ----
  gemm_wmma<false, false, true><<<gE, blk, 0, stream>>>(ah, tqT, tq_b, nullptr, q,  M, E, E);
  gemm_wmma<false, false, true><<<gE, blk, 0, stream>>>(ah, tkT, tk_b, nullptr, kk, M, E, E);
  gemm_wmma<false, false, true><<<gE, blk, 0, stream>>>(ah, tvT, tv_b, nullptr, vv, M, E, E);
  temporal_attn<<<dim3(1536), blk, 0, stream>>>(q, kk, vv, attn);
  gemm_wmma<false, true, false><<<gE, blk, 0, stream>>>(attn, toT, to_b, x1, tmp, M, E, E);
  ln_kernel<true><<<dim3(16384), blk, 0, stream>>>(tmp, tg, tb, x2, ah);

  // ---- FFN ----
  gemm_wmma<true, false, true><<<gF, blk, 0, stream>>>(ah, w1T, b1, nullptr, hbuf, M, DFF, E);
  gemm_wmma<false, true, false><<<gE, blk, 0, stream>>>(hbuf, w2T, b2, x2, tmp, M, E, DFF);
  ln_kernel<false><<<dim3(16384), blk, 0, stream>>>(tmp, g, bb, (float*)d_out, nullptr);
}
